// TopkGate_28793460752946
// MI455X (gfx1250) — matrix operand order repaired and checked by reading, not compiled, not run on hardware
//
#include <hip/hip_runtime.h>

// ---------------- MoE top-2 gate for MI455X (gfx1250) ----------------
// logits = x @ W^T + b ; softmax ; top-2 scatter.
// Memory-bound on streaming x (134 MB) -> design streams x exactly once,
// converts f32->bf16 in-register, and uses v_wmma_f32_16x16x32_bf16.
// W (512 KB f32) is staged once per workgroup as bf16 in LDS (264 KB of the
// 320 KB/WGP pool), bank-conflict-free via a 32 B row pad.

typedef __attribute__((ext_vector_type(16))) __bf16 v16bf;
typedef __attribute__((ext_vector_type(8)))  float  v8f;
typedef __attribute__((ext_vector_type(4)))  float  v4f;

#define C_IN            2048
#define N_EXP           64
#define KSTEPS          (C_IN / 32)     // 64 WMMA k-steps
#define WAVES_PER_BLOCK 8
#define TOKENS_PER_WAVE 16
#define TOKENS_PER_BLOCK (WAVES_PER_BLOCK * TOKENS_PER_WAVE)  // 128
#define W_ROW_STRIDE    2064            // bf16 elements per expert row (2048 + 16 pad)
#define SMEM_BYTES      (N_EXP * W_ROW_STRIDE * 2)  // 264192 B
#define LOGIT_STRIDE    68              // floats per token row in epilogue LDS

__launch_bounds__(256, 1)
__global__ void topk_gate_wmma(const float* __restrict__ x,
                               const float* __restrict__ W,
                               const float* __restrict__ b,
                               float* __restrict__ out)
{
    extern __shared__ char smem[];
    __bf16* wB = (__bf16*)smem;

    const int tid  = threadIdx.x;
    const int lane = tid & 31;
    const int wv   = tid >> 5;
    const int lm   = lane & 15;
    const int lh   = lane >> 4;          // 0: lanes 0-15, 1: lanes 16-31

    // ---- Phase 0: stage W as bf16 in LDS (coalesced f32x2 loads) ----
    {
        const float2* W2 = (const float2*)W;
        for (int i = tid; i < N_EXP * (C_IN / 2); i += 256) {
            const int e  = i >> 10;            // / (C_IN/2)
            const int k2 = (i & 1023) << 1;
            float2 w = W2[i];
            union { __bf16 h[2]; unsigned u; } p;
            p.h[0] = (__bf16)w.x;
            p.h[1] = (__bf16)w.y;
            *(unsigned*)&wB[e * W_ROW_STRIDE + k2] = p.u;
        }
    }
    __syncthreads();

    // ---- Phase 1: GEMM  M=16 tokens/wave, N=64 experts, K=2048 ----
    const long tokenBase = (long)blockIdx.x * TOKENS_PER_BLOCK + wv * TOKENS_PER_WAVE;
    const float* xrow = x + (tokenBase + lm) * C_IN;

    v8f acc[4] = {};

    for (int kc = 0; kc < KSTEPS; ++kc) {
        // A fragment, contiguous k order (k = 16*lh + s), same as B:
        // lane lm (lh=0): K = kc*32 + {0..15}
        // lane lm+16    : K = kc*32 + {16..31}
        const int c0 = kc * 32 + lh * 16;
        v4f a0 = *(const v4f*)(xrow + c0);
        v4f a1 = *(const v4f*)(xrow + c0 + 4);
        v4f a2 = *(const v4f*)(xrow + c0 + 8);
        v4f a3 = *(const v4f*)(xrow + c0 + 12);
        __builtin_prefetch(xrow + c0 + 128, 0, 0);   // global_prefetch_b8 down the stream

        v16bf A;
        #pragma unroll
        for (int j = 0; j < 4; ++j) {
            A[j]      = (__bf16)a0[j];
            A[4 + j]  = (__bf16)a1[j];
            A[8 + j]  = (__bf16)a2[j];
            A[12 + j] = (__bf16)a3[j];
        }

        // B fragments, contiguous k order matching A: N = lane&15,
        // lanes 0-15 hold K 0..15, lanes 16-31 hold K 16..31 (contiguous 32 B).
        const __bf16* wk = wB + kc * 32 + lh * 16;
        #pragma unroll
        for (int n = 0; n < 4; ++n) {
            v16bf Bf = *(const v16bf*)(wk + (n * 16 + lm) * W_ROW_STRIDE);
            acc[n] = __builtin_amdgcn_wmma_f32_16x16x32_bf16(
                false, A, false, Bf, (short)0, acc[n], false, false);
        }
    }

    // ---- Phase 2: bias (C/D layout: N = lane&15 for both lane halves) ----
    #pragma unroll
    for (int n = 0; n < 4; ++n) {
        const float bn = b[n * 16 + lm];
        #pragma unroll
        for (int j = 0; j < 8; ++j) acc[n][j] += bn;
    }

    // ---- Phase 3: park logits in LDS (W staging no longer needed) ----
    __syncthreads();                        // everyone done reading wB
    float* lg = (float*)smem + wv * (TOKENS_PER_WAVE * LOGIT_STRIDE);
    #pragma unroll
    for (int n = 0; n < 4; ++n)
        #pragma unroll
        for (int j = 0; j < 8; ++j)
            lg[(j + lh * 8) * LOGIT_STRIDE + (n * 16 + lm)] = acc[n][j];
    __syncthreads();

    // ---- Phase 4: softmax + top-2, two lanes per token ----
    // lane lm handles experts [lh*32, lh*32+32) of token (tokenBase + lm).
    const float* L = lg + lm * LOGIT_STRIDE + lh * 32;

    float mx = -3.4e38f;
    #pragma unroll
    for (int q = 0; q < 8; ++q) {
        v4f v = *(const v4f*)(L + q * 4);
        mx = fmaxf(mx, fmaxf(fmaxf(v[0], v[1]), fmaxf(v[2], v[3])));
    }
    const float M = fmaxf(mx, __shfl_xor(mx, 16, 32));

    float v1 = -3.4e38f, v2 = -3.4e38f, s = 0.f;
    int   i1 = -1,       i2 = -1;
    #pragma unroll
    for (int q = 0; q < 8; ++q) {
        v4f v = *(const v4f*)(L + q * 4);
        #pragma unroll
        for (int j = 0; j < 4; ++j) {
            const float l = v[j];
            const int   e = lh * 32 + q * 4 + j;
            s += __expf(l - M);
            if (l > v1)      { v2 = v1; i2 = i1; v1 = l; i1 = e; }
            else if (l > v2) { v2 = l;  i2 = e; }
        }
    }
    const float sTot = s + __shfl_xor(s, 16, 32);
    const float ov1 = __shfl_xor(v1, 16, 32); const int oi1 = __shfl_xor(i1, 16, 32);
    const float ov2 = __shfl_xor(v2, 16, 32); const int oi2 = __shfl_xor(i2, 16, 32);

    // merge two sorted pairs; ties -> lower expert index (matches lax.top_k)
    const bool  fm  = (v1 > ov1) || (v1 == ov1 && i1 < oi1);
    const float b1v = fm ? v1 : ov1;  const int b1i = fm ? i1 : oi1;
    const float cav = fm ? v2 : ov2;  const int cai = fm ? i2 : oi2;  // runner-up, winner side
    const float cbv = fm ? ov1 : v1;  const int cbi = fm ? oi1 : i1;  // best of loser side
    const bool  sa  = (cav > cbv) || (cav == cbv && cai < cbi);
    const float b2v = sa ? cav : cbv; const int b2i = sa ? cai : cbi;

    const float p1 = __expf(b1v - M) / sTot;
    const float p2 = __expf(b2v - M) / sTot;

    float* orow = out + (tokenBase + lm) * N_EXP + lh * 32;
    #pragma unroll
    for (int q = 0; q < 8; ++q) {
        v4f o;
        #pragma unroll
        for (int j = 0; j < 4; ++j) {
            const int e = lh * 32 + q * 4 + j;
            o[j] = (e == b1i) ? p1 : ((e == b2i) ? p2 : 0.f);
        }
        *(v4f*)(orow + q * 4) = o;
    }
}

extern "C" void kernel_launch(void* const* d_in, const int* in_sizes, int n_in,
                              void* d_out, int out_size, void* d_ws, size_t ws_size,
                              hipStream_t stream) {
    const float* x = (const float*)d_in[0];   // [4,4096,2048] f32
    const float* W = (const float*)d_in[1];   // [64,2048] f32
    const float* b = (const float*)d_in[2];   // [64] f32
    float* out = (float*)d_out;               // [4,4096,64] f32

    const int tokens = in_sizes[0] / C_IN;            // 16384
    const int grid   = tokens / TOKENS_PER_BLOCK;     // 128 workgroups
    topk_gate_wmma<<<grid, 256, SMEM_BYTES, stream>>>(x, W, b, out);
}
